// GAE_8753143349903
// MI455X (gfx1250) — matrix operand order repaired and checked by reading, not compiled, not run on hardware
//
#include <hip/hip_runtime.h>
#include <hip/hip_bf16.h>

#define NNODES 8192
#define NEDGES 262144

typedef __attribute__((ext_vector_type(16))) __bf16 v16bf;
typedef __attribute__((ext_vector_type(8)))  float  v8f;

__device__ __forceinline__ unsigned short f32_to_bf16_rn(float f) {
  unsigned int x = __float_as_uint(f);
  unsigned int lsb = (x >> 16) & 1u;
  x += 0x7fffu + lsb;                 // round-to-nearest-even
  return (unsigned short)(x >> 16);
}

__global__ void zero_f32_kernel(float* __restrict__ p, int n) {
  int i = blockIdx.x * blockDim.x + threadIdx.x;
  if (i < n) p[i] = 0.0f;
}

__global__ void deg_accum_kernel(const int* __restrict__ src, const int* __restrict__ dst,
                                 float* __restrict__ deg_out, float* __restrict__ deg_in, int e) {
  int i = blockIdx.x * blockDim.x + threadIdx.x;
  if (i < e) {
    unsafeAtomicAdd(&deg_out[src[i]], 1.0f);
    unsafeAtomicAdd(&deg_in[dst[i]], 1.0f);
  }
}

__global__ void deg_to_norm_kernel(float* __restrict__ p, int n) {
  int i = blockIdx.x * blockDim.x + threadIdx.x;
  if (i < n) p[i] = rsqrtf(fmaxf(p[i], 1.0f));
}

// xs = bf16(x * norm_src[row])
__global__ void prep_input_kernel(const float* __restrict__ x, const float* __restrict__ ns,
                                  unsigned short* __restrict__ xs, int n, int f) {
  int i = blockIdx.x * blockDim.x + threadIdx.x;
  if (i < n * f) xs[i] = f32_to_bf16_rn(x[i] * ns[i / f]);
}

// Wt[o*K + k] = bf16(W[k*O + o])   (W is [K,O] row-major)
__global__ void transpose_w_kernel(const float* __restrict__ w, unsigned short* __restrict__ wt,
                                   int k, int o) {
  int i = blockIdx.x * blockDim.x + threadIdx.x;
  if (i < k * o) {
    int kk = i / o, oo = i - kk * o;
    wt[oo * k + kk] = f32_to_bf16_rn(w[i]);
  }
}

// Generic layer GEMM: C[M,Nc] = A[M,K] * Bt[Nc,K]^T, one 16x16 tile per wave.
__global__ __launch_bounds__(256)
void gemm_bf16_wmma_kernel(const unsigned short* __restrict__ A,
                           const unsigned short* __restrict__ Bt,
                           float* __restrict__ C,
                           int M, int Nc, int K) {
  const int wave = threadIdx.x >> 5;
  const int lane = threadIdx.x & 31;
  const int tiles_n = Nc >> 4;
  const long total  = (long)(M >> 4) * tiles_n;
  const long tile   = (long)blockIdx.x * 8 + wave;
  if (tile >= total) return;               // wave-uniform: EXEC all-ones inside
  const int tm = (int)(tile / tiles_n);
  const int tn = (int)(tile % tiles_n);
  const int r  = lane & 15;
  const int hi = lane >> 4;

  const unsigned short* arow = A  + (size_t)(tm * 16 + r) * K + hi * 8;
  const unsigned short* brow = Bt + (size_t)(tn * 16 + r) * K + hi * 8;

  v8f acc = {0.f, 0.f, 0.f, 0.f, 0.f, 0.f, 0.f, 0.f};
  for (int k0 = 0; k0 < K; k0 += 32) {
    union { v16bf v; uint4 q[2]; } a, b;
    a.q[0] = *(const uint4*)(arow + k0);
    a.q[1] = *(const uint4*)(arow + k0 + 16);
    b.q[0] = *(const uint4*)(brow + k0);
    b.q[1] = *(const uint4*)(brow + k0 + 16);
    acc = __builtin_amdgcn_wmma_f32_16x16x32_bf16(false, a.v, false, b.v,
                                                  (short)0, acc, false, false);
  }

  float* crow = C + (size_t)(tm * 16 + hi * 8) * Nc + (size_t)tn * 16 + r;
#pragma unroll
  for (int v = 0; v < 8; ++v) crow[(size_t)v * Nc] = acc[v];
}

// adj kernel: C[n,n] = sigmoid(Z * Z^T), K = 128 fixed.
// One 16x64 output strip per wave: the whole 16x128 A strip lives in VGPRs
// (4 frags) and is reused across 4 N-subtiles -> 16 v_wmma per strip,
// 1.6x less L2 read traffic than one-tile-per-wave.
__global__ __launch_bounds__(256)
void adj_sigmoid_wmma_kernel(const unsigned short* __restrict__ Z,
                             float* __restrict__ C, int n) {
  const int wave = threadIdx.x >> 5;
  const int lane = threadIdx.x & 31;
  const int strips_n = n >> 6;                         // 64-wide strips
  const long total   = (long)(n >> 4) * strips_n;
  const long strip   = (long)blockIdx.x * 8 + wave;
  if (strip >= total) return;                          // wave-uniform
  const int tm = (int)(strip / strips_n);
  const int sn = (int)(strip % strips_n);
  const int r  = lane & 15;
  const int hi = lane >> 4;

  union F { v16bf v; uint4 q[2]; };

  // preload full A strip (K = 0..127) into registers
  const unsigned short* arow = Z + (size_t)(tm * 16 + r) * 128 + hi * 8;
  F a[4];
#pragma unroll
  for (int kk = 0; kk < 4; ++kk) {
    a[kk].q[0] = *(const uint4*)(arow + kk * 32);
    a[kk].q[1] = *(const uint4*)(arow + kk * 32 + 16);
  }

  v8f acc[4];
#pragma unroll
  for (int t = 0; t < 4; ++t)
    acc[t] = (v8f){0.f, 0.f, 0.f, 0.f, 0.f, 0.f, 0.f, 0.f};

#pragma unroll
  for (int t = 0; t < 4; ++t) {
    const unsigned short* brow = Z + (size_t)(sn * 64 + t * 16 + r) * 128 + hi * 8;
#pragma unroll
    for (int kk = 0; kk < 4; ++kk) {
      F b;
      b.q[0] = *(const uint4*)(brow + kk * 32);
      b.q[1] = *(const uint4*)(brow + kk * 32 + 16);
      acc[t] = __builtin_amdgcn_wmma_f32_16x16x32_bf16(false, a[kk].v, false, b.v,
                                                       (short)0, acc[t], false, false);
    }
  }

#pragma unroll
  for (int t = 0; t < 4; ++t) {
    float* crow = C + (size_t)(tm * 16 + hi * 8) * n + (size_t)(sn * 64 + t * 16) + r;
#pragma unroll
    for (int v = 0; v < 8; ++v)
      crow[(size_t)v * n] = 1.0f / (1.0f + __expf(-acc[t][v]));
  }
}

// agg[dst[e], :] += m[src[e], :]   (one wave32 per edge; hw f32 atomics)
__global__ void scatter_kernel(const float* __restrict__ m, const int* __restrict__ src,
                               const int* __restrict__ dst, float* __restrict__ agg,
                               int e, int f) {
  int gw   = (blockIdx.x * blockDim.x + threadIdx.x) >> 5;
  int lane = threadIdx.x & 31;
  if (gw >= e) return;
  const float* mr = m   + (size_t)src[gw] * f;
  float*       ar = agg + (size_t)dst[gw] * f;
  for (int j = lane; j < f; j += 32)
    unsafeAtomicAdd(&ar[j], mr[j]);
}

// h = agg*norm_dst + b; optional relu; optional f32 out; optional bf16 next-input
__global__ void finalize_kernel(const float* __restrict__ agg, const float* __restrict__ nd,
                                const float* __restrict__ bias, int n, int f, int relu,
                                float* __restrict__ fout, unsigned short* __restrict__ xsout,
                                const float* __restrict__ nsrc) {
  int i = blockIdx.x * blockDim.x + threadIdx.x;
  if (i >= n * f) return;
  int row = i / f, col = i - row * f;
  float h = agg[i] * nd[row] + bias[col];
  if (relu) h = fmaxf(h, 0.0f);
  if (fout) fout[i] = h;
  if (xsout) {
    float s = nsrc ? nsrc[row] : 1.0f;
    xsout[i] = f32_to_bf16_rn(h * s);
  }
}

extern "C" void kernel_launch(void* const* d_in, const int* in_sizes, int n_in,
                              void* d_out, int out_size, void* d_ws, size_t ws_size,
                              hipStream_t stream) {
  (void)in_sizes; (void)n_in; (void)out_size; (void)ws_size;
  const int N = NNODES, E = NEDGES, T = 256;

  const float* features = (const float*)d_in[0];
  const int*   src      = (const int*)d_in[1];
  const int*   dst      = (const int*)d_in[2];

  // workspace layout (~13.1 MB), 16B-aligned sub-buffers
  float*          norm_src = (float*)d_ws;                       // N
  float*          norm_dst = norm_src + N;                       // N
  unsigned short* Wt       = (unsigned short*)(norm_dst + N);    // 128*128
  unsigned short* xs       = Wt + 128 * 128;                     // N*128 bf16
  unsigned short* zs       = xs + (size_t)N * 128;               // N*64  bf16 (scaled z)
  unsigned short* zb       = zs + (size_t)N * 64;                // N*128 bf16 (z_ for adj)
  float*          mbuf     = (float*)(zb + (size_t)N * 128);     // N*128 f32
  float*          agg      = mbuf + (size_t)N * 128;             // N*128 f32

  float* out_z     = (float*)d_out;                  // [N,64]
  float* out_recon = out_z + (size_t)N * 64;         // [N,128]
  float* out_adj   = out_recon + (size_t)N * 128;    // [N,N]

  auto nb = [](long n, int t) { return (unsigned)((n + t - 1) / t); };

  // degree norms (norm_src/norm_dst are contiguous -> one pass over 2N)
  zero_f32_kernel<<<nb(2L * N, T), T, 0, stream>>>(norm_src, 2 * N);
  deg_accum_kernel<<<nb(E, T), T, 0, stream>>>(src, dst, norm_src, norm_dst, E);
  deg_to_norm_kernel<<<nb(2L * N, T), T, 0, stream>>>(norm_src, 2 * N);

  prep_input_kernel<<<nb((long)N * 128, T), T, 0, stream>>>(features, norm_src, xs, N, 128);

  auto layer = [&](const unsigned short* in_xs, const float* W, const float* b,
                   int K, int O, int relu, float* fout, unsigned short* xsout,
                   const float* nsrc) {
    transpose_w_kernel<<<nb((long)K * O, T), T, 0, stream>>>(W, Wt, K, O);
    long tiles = (long)(N / 16) * (O / 16);
    gemm_bf16_wmma_kernel<<<nb(tiles, 8), T, 0, stream>>>(in_xs, Wt, mbuf, N, O, K);
    zero_f32_kernel<<<nb((long)N * O, T), T, 0, stream>>>(agg, N * O);
    scatter_kernel<<<nb((long)E * 32, T), T, 0, stream>>>(mbuf, src, dst, agg, E, O);
    finalize_kernel<<<nb((long)N * O, T), T, 0, stream>>>(agg, norm_dst, b, N, O, relu,
                                                          fout, xsout, nsrc);
  };

  // encoder: 128 -> 128 -> 128 -> 64 (z)
  layer(xs, (const float*)d_in[3],  (const float*)d_in[4],  128, 128, 1, nullptr,   xs, norm_src);
  layer(xs, (const float*)d_in[5],  (const float*)d_in[6],  128, 128, 1, nullptr,   xs, norm_src);
  layer(xs, (const float*)d_in[7],  (const float*)d_in[8],  128,  64, 0, out_z,     zs, norm_src);
  // A-decoder: 64 -> 128 -> 128 -> 128 (recon)
  layer(zs, (const float*)d_in[9],  (const float*)d_in[10],  64, 128, 1, nullptr,   xs, norm_src);
  layer(xs, (const float*)d_in[11], (const float*)d_in[12], 128, 128, 1, nullptr,   xs, norm_src);
  layer(xs, (const float*)d_in[13], (const float*)d_in[14], 128, 128, 0, out_recon, nullptr, nullptr);
  // S-decoder: 64 -> 128 -> 128 (z_)
  layer(zs, (const float*)d_in[15], (const float*)d_in[16],  64, 128, 1, nullptr,   xs, norm_src);
  layer(xs, (const float*)d_in[17], (const float*)d_in[18], 128, 128, 0, nullptr,   zb, nullptr);

  // adj = sigmoid(z_ @ z_^T): 16x64 strip per wave, sigmoid fused
  long strips = (long)(N / 16) * (N / 64);
  adj_sigmoid_wmma_kernel<<<nb(strips, 8), T, 0, stream>>>(zb, out_adj, N);
}
